// FineTuneMolFormer_21337397527100
// MI455X (gfx1250) — matrix operand order repaired and checked by reading, not compiled, not run on hardware
//
#include <hip/hip_runtime.h>
#include <hip/hip_bf16.h>

#define DEV __device__ __forceinline__

namespace mf {

constexpr int Bc = 16, Lc = 2048, Dc = 768, Hc = 12, NLc = 4, Fc = 32;
constexpr int HDc = Dc / Hc;                 // 64
constexpr int BL  = Bc * Lc;                 // 32768
constexpr long BLD = (long)BL * Dc;          // 25,165,824 elements
constexpr long PHN = (long)BL * Hc * Fc;     // 12,582,912 elements
constexpr long WSZ = (long)Dc * Dc;          // 589,824 weight elements

typedef __attribute__((ext_vector_type(16))) __bf16 v16bf;
typedef __attribute__((ext_vector_type(8)))  __bf16 v8bf;
typedef __attribute__((ext_vector_type(8)))  float  v8f;

struct BF16x16 { v8bf lo, hi; };

DEV __bf16 f2bf(float f) {
  unsigned u = __builtin_bit_cast(unsigned, f);
  unsigned r = u + 0x7fffu + ((u >> 16) & 1u);   // round-to-nearest-even
  unsigned short hv = (unsigned short)(r >> 16);
  return __builtin_bit_cast(__bf16, hv);
}

DEV float gelu_exact(float x) { return 0.5f * x * (1.0f + erff(x * 0.7071067811865475f)); }

enum { EPI_BIAS = 0, EPI_GELU = 1, EPI_RES = 2 };

// C[M,N] = epi( A[M,K](bf16) @ Bt[N,K](bf16, pre-transposed) + bias (+ res) )
// WG tile 128x128, BK=32; 8 waves x (32x64) each.
// A tile staged to LDS with GLOBAL_LOAD_ASYNC_TO_LDS_B128 (ASYNCcnt);
// B fragments read directly from L2-resident transposed weights.
template <int EPI, bool OUTBF>
__global__ __launch_bounds__(256) void gemm_bf16_wmma(
    const __bf16* __restrict__ A, const __bf16* __restrict__ Bt,
    const float* __restrict__ bias, const float* __restrict__ res,
    float* __restrict__ C, __bf16* __restrict__ Cb, int M, int N, int K) {
  constexpr int LDA = 40;                    // halves per LDS row (32 + 8 pad)
  __shared__ __align__(16) __bf16 sA[128 * LDA];   // only shared object -> LDS offset 0

  const int bm = blockIdx.x * 128;
  const int bn = blockIdx.y * 128;
  const int tid  = threadIdx.x;
  const int lane = tid & 31;
  const int wave = tid >> 5;
  const int wm = wave & 3;    // M sub-tile (32 rows each)
  const int wn = wave >> 2;   // N sub-tile (64 cols each)
  const int hi = lane >> 4;   // lane half
  const int ln = lane & 15;

  // Per-lane B row pointers (Bt row-major [N,K]); v8bf element = 16 bytes.
  const v8bf* brow[4];
#pragma unroll
  for (int tj = 0; tj < 4; ++tj)
    brow[tj] = (const v8bf*)(Bt + (size_t)(bn + wn * 64 + tj * 16 + ln) * K);

  v8f acc[2][4];
#pragma unroll
  for (int ti = 0; ti < 2; ++ti)
#pragma unroll
    for (int tj = 0; tj < 4; ++tj) acc[ti][tj] = {};

  for (int kk = 0; kk < K; kk += 32) {
    // ---- async-copy A tile: 128 rows x 32 bf16 = 512 x 16B chunks ----
#pragma unroll
    for (int r = 0; r < 2; ++r) {
      int pos = tid + 256 * r;
      int row = pos >> 2;               // 4 chunks per row
      int c16 = pos & 3;
      const __bf16* gsrc = A + (size_t)(bm + row) * K + kk + c16 * 8;
      unsigned ldsoff = (unsigned)(row * (LDA * 2) + c16 * 16);   // bytes, sA @ 0
      asm volatile("global_load_async_to_lds_b128 %0, %1, off"
                   :: "v"(ldsoff), "v"(gsrc) : "memory");
    }
    if (kk + 32 < K) {   // prefetch next A tile rows (global_prefetch_b8)
      __builtin_prefetch(A + (size_t)(bm + (tid >> 1)) * K + kk + 32 + (tid & 1) * 16, 0, 1);
    }
    asm volatile("s_wait_asynccnt 0x0" ::: "memory");
    __syncthreads();

    // ---- A fragments from LDS (ISA 7.12.2 16-bit A layout) ----
    v16bf afrag[2];
#pragma unroll
    for (int ti = 0; ti < 2; ++ti) {
      int m = wm * 32 + ti * 16 + ln;
      BF16x16 t;
      t.lo = *(const v8bf*)&sA[m * LDA + hi * 8];
      t.hi = *(const v8bf*)&sA[m * LDA + 16 + hi * 8];
      afrag[ti] = __builtin_bit_cast(v16bf, t);
    }
    // ---- B fragments direct from global (col n, K = hi*8..+7 and 16+hi*8..+7) ----
    int boff = (kk >> 3) + hi;
    v16bf bfrag[4];
#pragma unroll
    for (int tj = 0; tj < 4; ++tj) {
      BF16x16 t;
      t.lo = brow[tj][boff];
      t.hi = brow[tj][boff + 2];
      bfrag[tj] = __builtin_bit_cast(v16bf, t);
    }
#pragma unroll
    for (int ti = 0; ti < 2; ++ti)
#pragma unroll
      for (int tj = 0; tj < 4; ++tj)
        acc[ti][tj] = __builtin_amdgcn_wmma_f32_16x16x32_bf16(
            false, afrag[ti], false, bfrag[tj], (short)0, acc[ti][tj], false, false);
    __syncthreads();
  }

  // ---- epilogue: C/D layout: VGPR j, lane -> m = hi*8 + j, n = ln ----
#pragma unroll
  for (int ti = 0; ti < 2; ++ti)
#pragma unroll
    for (int tj = 0; tj < 4; ++tj)
#pragma unroll
      for (int j = 0; j < 8; ++j) {
        int m = bm + wm * 32 + ti * 16 + hi * 8 + j;
        int n = bn + wn * 64 + tj * 16 + ln;
        float v = acc[ti][tj][j] + bias[n];
        if (EPI == EPI_GELU) v = gelu_exact(v);
        if (EPI == EPI_RES)  v += res[(size_t)m * N + n];
        if (OUTBF) Cb[(size_t)m * N + n] = f2bf(v);
        else       C[(size_t)m * N + n] = v;
      }
}

// Weight transpose + fp32->bf16: w[K,N] -> wt[N,K]
__global__ __launch_bounds__(256) void wtrans_kernel(const float* __restrict__ w,
                                                     __bf16* __restrict__ wt,
                                                     int K, int N) {
  __shared__ float tile[32][33];
  int k0 = blockIdx.x * 32, n0 = blockIdx.y * 32;
  int tx = threadIdx.x & 31, ty = threadIdx.x >> 5;   // 32 x 8
#pragma unroll
  for (int r = 0; r < 32; r += 8)
    tile[ty + r][tx] = w[(size_t)(k0 + ty + r) * N + n0 + tx];
  __syncthreads();
#pragma unroll
  for (int r = 0; r < 32; r += 8)
    wt[(size_t)(n0 + ty + r) * K + k0 + tx] = f2bf(tile[tx][ty + r]);
}

// x[row,:] = tok_emb[idx[row],:]  (fp32 + bf16 mirror)
__global__ void embed_kernel(const int* __restrict__ idx,
                             const float* __restrict__ emb,
                             float* __restrict__ x, __bf16* __restrict__ xb) {
  int row = blockIdx.x;
  const float* src = emb + (size_t)idx[row] * Dc;
  float* dst = x + (size_t)row * Dc;
  __bf16* db = xb + (size_t)row * Dc;
  for (int d = threadIdx.x; d < Dc; d += blockDim.x) {
    float v = src[d];
    dst[d] = v;
    db[d] = f2bf(v);
  }
}

// In-place rotary on q and k. One thread per (row, i) pair, i in [0,32).
__global__ void rotary_kernel(float* __restrict__ q, float* __restrict__ k) {
  long e = (long)blockIdx.x * 256 + threadIdx.x;   // B*L*H*32 exact
  int i = (int)(e & 31);
  long r = e >> 5;                // (b*L + l)*H + h
  long bl = r / Hc;
  int l = (int)(bl % Lc);
  float inv = exp2f(-(float)i * (13.287712379549449f / 32.0f));  // 10000^(-i/32)
  float ang = (float)l * inv;
  float c = cosf(ang), s = sinf(ang);
  size_t base = (size_t)r * HDc;
  float q1 = q[base + i], q2 = q[base + i + 32];
  q[base + i]      = q1 * c - q2 * s;
  q[base + i + 32] = q2 * c + q1 * s;
  float k1 = k[base + i], k2 = k[base + i + 32];
  k[base + i]      = k1 * c - k2 * s;
  k[base + i + 32] = k2 * c + k1 * s;
}

// dst[r,f] = relu(sum_d src[r,d]*omega[d,f]) (* mask)
__global__ void phi_kernel(const float* __restrict__ src,
                           const float* __restrict__ omega,
                           const int* __restrict__ mask,
                           float* __restrict__ dst) {
  __shared__ float som[HDc * Fc];
  for (int i = threadIdx.x; i < HDc * Fc; i += 256) som[i] = omega[i];
  __syncthreads();
  long total = (long)BL * Hc * Fc;
  for (long o = (long)blockIdx.x * 256 + threadIdx.x; o < total;
       o += (long)gridDim.x * 256) {
    int f = (int)(o & 31);
    long r = o >> 5;
    const float* x = src + r * HDc;
    float a = 0.f;
#pragma unroll
    for (int d = 0; d < HDc; ++d) a += x[d] * som[d * Fc + f];
    a = a > 0.f ? a : 0.f;
    if (mask) a *= (float)mask[r / Hc];
    dst[o] = a;
  }
}

// Per (b,h): kv[f,d] = sum_l phk[b,l,h,f]*v[b,l,h,d]; s[f] = sum_l phk
__global__ __launch_bounds__(256) void kv_kernel(
    const float* __restrict__ phk, const float* __restrict__ v,
    float* __restrict__ kvb, float* __restrict__ sb) {
  int bh = blockIdx.x;
  int b = bh / Hc, h = bh % Hc;
  int t = threadIdx.x;
  int f = t >> 3;        // 0..31
  int dg = t & 7;        // d = dg + 8*j
  float acc[8] = {0, 0, 0, 0, 0, 0, 0, 0};
  float sacc = 0.f;
  const float* pk = phk + ((size_t)b * Lc * Hc + h) * Fc + f;
  const float* pv = v + (size_t)b * Lc * Dc + h * HDc + dg;
  for (int l = 0; l < Lc; ++l) {
    float ph = pk[(size_t)l * Hc * Fc];
    const float* pvl = pv + (size_t)l * Dc;
#pragma unroll
    for (int j = 0; j < 8; ++j) acc[j] += ph * pvl[8 * j];
    sacc += ph;
  }
#pragma unroll
  for (int j = 0; j < 8; ++j)
    kvb[((size_t)bh * Fc + f) * HDc + dg + 8 * j] = acc[j];
  if (dg == 0) sb[(size_t)bh * Fc + f] = sacc;
}

// att(bf16)[b,l,h,d] = (sum_f phq*kv[f,d]) / (sum_f phq*s[f] + eps)
__global__ __launch_bounds__(256) void attnout_kernel(
    const float* __restrict__ phq, const float* __restrict__ kvb,
    const float* __restrict__ sb, __bf16* __restrict__ att) {
  __shared__ float skv[Fc * HDc];
  __shared__ float ss[Fc];
  int bh = blockIdx.x;
  int b = bh / Hc, h = bh % Hc;
  for (int i = threadIdx.x; i < Fc * HDc; i += 256)
    skv[i] = kvb[(size_t)bh * Fc * HDc + i];
  if (threadIdx.x < Fc) ss[threadIdx.x] = sb[(size_t)bh * Fc + threadIdx.x];
  __syncthreads();
  int l = blockIdx.y * 4 + (threadIdx.x >> 6);
  int d = threadIdx.x & 63;
  const float* pq = phq + ((size_t)(b * Lc + l) * Hc + h) * Fc;
  float num = 0.f, den = 0.f;
#pragma unroll
  for (int f = 0; f < Fc; ++f) {
    float p = pq[f];
    num += p * skv[f * HDc + d];
    den += p * ss[f];
  }
  att[(size_t)(b * Lc + l) * Dc + h * HDc + d] = f2bf(num / (den + 1e-6f));
}

// Row layernorm over D=768, eps=1e-5; fp32 out + optional bf16 mirror
__global__ __launch_bounds__(256) void ln_kernel(
    const float* __restrict__ in, const float* __restrict__ g,
    const float* __restrict__ bb, float* __restrict__ out,
    __bf16* __restrict__ outb) {
  __shared__ float red[256];
  int row = blockIdx.x, tid = threadIdx.x;
  const float* x = in + (size_t)row * Dc;
  float s = 0.f, ss = 0.f;
  for (int d = tid; d < Dc; d += 256) { float v = x[d]; s += v; ss += v * v; }
  red[tid] = s; __syncthreads();
  for (int o = 128; o > 0; o >>= 1) { if (tid < o) red[tid] += red[tid + o]; __syncthreads(); }
  float mean = red[0] / Dc; __syncthreads();
  red[tid] = ss; __syncthreads();
  for (int o = 128; o > 0; o >>= 1) { if (tid < o) red[tid] += red[tid + o]; __syncthreads(); }
  float var = red[0] / Dc - mean * mean;
  float inv = rsqrtf(var + 1e-5f);
  float* y = out + (size_t)row * Dc;
  __bf16* yb = outb ? outb + (size_t)row * Dc : nullptr;
  for (int d = tid; d < Dc; d += 256) {
    float v = (x[d] - mean) * inv * g[d] + bb[d];
    y[d] = v;
    if (yb) yb[d] = f2bf(v);
  }
}

}  // namespace mf

extern "C" void kernel_launch(void* const* d_in, const int* in_sizes, int n_in,
                              void* d_out, int out_size, void* d_ws, size_t ws_size,
                              hipStream_t stream) {
  using namespace mf;
  const int*   idx  = (const int*)d_in[0];
  const int*   mask = (const int*)d_in[1];
  const float* tok  = (const float*)d_in[2];
  const float* Wq   = (const float*)d_in[3];
  const float* bq   = (const float*)d_in[4];
  const float* Wk   = (const float*)d_in[5];
  const float* bk   = (const float*)d_in[6];
  const float* Wv   = (const float*)d_in[7];
  const float* bv   = (const float*)d_in[8];
  const float* Wo   = (const float*)d_in[9];
  const float* bo   = (const float*)d_in[10];
  const float* omg  = (const float*)d_in[11];
  const float* W1   = (const float*)d_in[12];
  const float* b1   = (const float*)d_in[13];
  const float* W2   = (const float*)d_in[14];
  const float* b2   = (const float*)d_in[15];
  const float* g1   = (const float*)d_in[16];
  const float* be1  = (const float*)d_in[17];
  const float* g2   = (const float*)d_in[18];
  const float* be2  = (const float*)d_in[19];
  const float* gf   = (const float*)d_in[20];
  const float* bf   = (const float*)d_in[21];
  float* outp = (float*)d_out;

  // ---- workspace carve ----
  float* F0 = (float*)d_ws;
  float* X   = F0;              // activations [B,L,D] fp32
  float* Ab  = X + BLD;         // pre-LN residual sums
  float* Q   = Ab + BLD;
  float* Kb  = Q + BLD;
  float* V   = Kb + BLD;
  float* PHQ = V + BLD;         // [B,L,H,F]
  float* PHK = PHQ + PHN;
  float* KV  = PHK + PHN;       // [B,H,F,HD]
  float* Sb  = KV + (long)Bc * Hc * Fc * HDc;  // [B,H,F]
  __bf16* Hf = (__bf16*)(Sb + (long)Bc * Hc * Fc);
  __bf16* Xb  = Hf;             // bf16 mirror of X        [B,L,D]
  __bf16* Gb  = Xb + BLD;       // bf16 attn-out / ffn-hidden
  __bf16* WtQ = Gb + BLD;       // transposed bf16 weights [N,K]
  __bf16* WtK = WtQ + WSZ;
  __bf16* WtV = WtK + WSZ;
  __bf16* WtO = WtV + WSZ;
  __bf16* Wt1 = WtO + WSZ;
  __bf16* Wt2 = Wt1 + WSZ;

  const dim3 ggrid(BL / 128, Dc / 128);   // (256, 6)
  const dim3 tgrid(Dc / 32, Dc / 32);     // (24, 24)
  const dim3 blk(256);

  embed_kernel<<<BL, 256, 0, stream>>>(idx, tok, X, Xb);

  for (int i = 0; i < NLc; ++i) {
    wtrans_kernel<<<tgrid, blk, 0, stream>>>(Wq + (size_t)i * WSZ, WtQ, Dc, Dc);
    wtrans_kernel<<<tgrid, blk, 0, stream>>>(Wk + (size_t)i * WSZ, WtK, Dc, Dc);
    wtrans_kernel<<<tgrid, blk, 0, stream>>>(Wv + (size_t)i * WSZ, WtV, Dc, Dc);
    wtrans_kernel<<<tgrid, blk, 0, stream>>>(Wo + (size_t)i * WSZ, WtO, Dc, Dc);
    wtrans_kernel<<<tgrid, blk, 0, stream>>>(W1 + (size_t)i * WSZ, Wt1, Dc, Dc);
    wtrans_kernel<<<tgrid, blk, 0, stream>>>(W2 + (size_t)i * WSZ, Wt2, Dc, Dc);

    gemm_bf16_wmma<EPI_BIAS, false><<<ggrid, blk, 0, stream>>>(
        Xb, WtQ, bq + (size_t)i * Dc, nullptr, Q, nullptr, BL, Dc, Dc);
    gemm_bf16_wmma<EPI_BIAS, false><<<ggrid, blk, 0, stream>>>(
        Xb, WtK, bk + (size_t)i * Dc, nullptr, Kb, nullptr, BL, Dc, Dc);
    gemm_bf16_wmma<EPI_BIAS, false><<<ggrid, blk, 0, stream>>>(
        Xb, WtV, bv + (size_t)i * Dc, nullptr, V, nullptr, BL, Dc, Dc);

    rotary_kernel<<<(BL * Hc * 32) / 256, 256, 0, stream>>>(Q, Kb);

    phi_kernel<<<4096, 256, 0, stream>>>(Q, omg + (size_t)i * HDc * Fc, nullptr, PHQ);
    phi_kernel<<<4096, 256, 0, stream>>>(Kb, omg + (size_t)i * HDc * Fc, mask, PHK);

    kv_kernel<<<Bc * Hc, 256, 0, stream>>>(PHK, V, KV, Sb);
    attnout_kernel<<<dim3(Bc * Hc, Lc / 4), 256, 0, stream>>>(PHQ, KV, Sb, Gb);

    // Ab = att @ Wo + bo + X
    gemm_bf16_wmma<EPI_RES, false><<<ggrid, blk, 0, stream>>>(
        Gb, WtO, bo + (size_t)i * Dc, X, Ab, nullptr, BL, Dc, Dc);
    // X(+Xb) = LN1(Ab)
    ln_kernel<<<BL, 256, 0, stream>>>(Ab, g1 + (size_t)i * Dc, be1 + (size_t)i * Dc, X, Xb);
    // Gb = gelu(X @ W1 + b1)  (bf16 out)
    gemm_bf16_wmma<EPI_GELU, true><<<ggrid, blk, 0, stream>>>(
        Xb, Wt1, b1 + (size_t)i * Dc, nullptr, nullptr, Gb, BL, Dc, Dc);
    // Ab = Gb @ W2 + b2 + X
    gemm_bf16_wmma<EPI_RES, false><<<ggrid, blk, 0, stream>>>(
        Gb, Wt2, b2 + (size_t)i * Dc, X, Ab, nullptr, BL, Dc, Dc);
    // X(+Xb) = LN2(Ab)
    ln_kernel<<<BL, 256, 0, stream>>>(Ab, g2 + (size_t)i * Dc, be2 + (size_t)i * Dc, X, Xb);
  }

  ln_kernel<<<BL, 256, 0, stream>>>(X, gf, bf, outp, nullptr);
}
